// CellBERTModel_7352984010955
// MI455X (gfx1250) — matrix operand order repaired and checked by reading, not compiled, not run on hardware
//
#include <hip/hip_runtime.h>
#include <hip/hip_bf16.h>

typedef __attribute__((ext_vector_type(16))) _Float16 v16h;
typedef __attribute__((ext_vector_type(8)))  float    v8f;

union frag16 { v16h v; _Float16 h[16]; unsigned u[8]; uint4 q[2]; };

__device__ inline v8f wmma_f16(v16h a, v16h b, v8f c) {
  // D = A(16x32 f16) * B(32x16 f16) + C(16x16 f32)
  return __builtin_amdgcn_wmma_f32_16x16x32_f16(false, a, false, b, (short)0, c,
                                                false, false);
}

// convert two float4 (8 floats) into halves fr.h[o..o+7]
__device__ inline void cvt8(frag16& fr, int o, float4 a, float4 b) {
  fr.h[o + 0] = (_Float16)a.x; fr.h[o + 1] = (_Float16)a.y;
  fr.h[o + 2] = (_Float16)a.z; fr.h[o + 3] = (_Float16)a.w;
  fr.h[o + 4] = (_Float16)b.x; fr.h[o + 5] = (_Float16)b.y;
  fr.h[o + 6] = (_Float16)b.z; fr.h[o + 7] = (_Float16)b.w;
}

#define S_PAD   1056
#define S_VALID 1025
#define DM      256
#define LDA     40   // LDS stride in halves (80B rows -> 16B-aligned fragment groups)

// ---------------------------------------------------------------------------
// f32 -> f16 weight conversion
// ---------------------------------------------------------------------------
__global__ void cvt_f32_f16(const float* __restrict__ s, _Float16* __restrict__ d, int n) {
  int i = blockIdx.x * 256 + threadIdx.x;
  if (i < n) d[i] = (_Float16)s[i];
}

// ---------------------------------------------------------------------------
// C[M,N] f32 = A[M,K] f32 @ W[K,N] f16 + bias (+resid) (opt relu)
// block 256 = 8 waves; tile 128x32; K-step 32; each wave -> 16x32 strip.
// Fragments load via ds_load_b128; next-tile prefetch via global_prefetch.
// ---------------------------------------------------------------------------
__global__ __launch_bounds__(256) void gemm_rrr(
    const float* __restrict__ A, const _Float16* __restrict__ W,
    const float* __restrict__ bias, const float* __restrict__ resid,
    float* __restrict__ C, int M, int N, int K, int relu) {
  __shared__ alignas(16) _Float16 Ash[128 * LDA];   // [row][k]
  __shared__ alignas(16) _Float16 BshT[32 * LDA];   // transposed: [n][k]
  int tid = threadIdx.x;
  int lane = tid & 31, wid = tid >> 5;
  int hsel = lane >> 4, l15 = lane & 15;
  int bm = blockIdx.x * 128, bn = blockIdx.y * 32;
  v8f acc0 = {}, acc1 = {};
  int arow = tid >> 1, acol = (tid & 1) * 16;   // A stage: 128 rows x 32 cols
  int brow = tid >> 3, bcol = (tid & 7) * 4;    // W stage: brow=k (0..31), bcol=n

  for (int k0 = 0; k0 < K; k0 += 32) {
    // ---- stage A (f32 -> f16), vector loads + ds_store_b64 ----
    const float4* ap =
        reinterpret_cast<const float4*>(A + (size_t)(bm + arow) * K + k0 + acol);
#pragma unroll
    for (int j = 0; j < 4; ++j) {
      float4 f = ap[j];
      union { _Float16 h[4]; uint2 u; } pk;
      pk.h[0] = (_Float16)f.x; pk.h[1] = (_Float16)f.y;
      pk.h[2] = (_Float16)f.z; pk.h[3] = (_Float16)f.w;
      *reinterpret_cast<uint2*>(&Ash[arow * LDA + acol + j * 4]) = pk.u;
    }
    // ---- stage W transposed ----
    const _Float16* wp = W + (size_t)(k0 + brow) * N + bn + bcol;
    union { uint2 u; _Float16 h[4]; } wv;
    wv.u = *reinterpret_cast<const uint2*>(wp);
#pragma unroll
    for (int j = 0; j < 4; ++j) BshT[(bcol + j) * LDA + brow] = wv.h[j];
    // ---- prefetch next K-step tiles (global_prefetch_b8) ----
    if (k0 + 32 < K) {
      __builtin_prefetch((const void*)(ap + 8), 0, 3);
      __builtin_prefetch((const void*)(wp + (size_t)32 * N), 0, 3);
    }
    __syncthreads();

    // A fragment: two contiguous 8-half groups -> 2x ds_load_b128
    frag16 a;
    int ar = wid * 16 + l15;
    a.q[0] = *reinterpret_cast<const uint4*>(&Ash[ar * LDA + hsel * 16]);
    a.q[1] = *reinterpret_cast<const uint4*>(&Ash[ar * LDA + hsel * 16 + 8]);
    // B fragments: 16 contiguous K-halves per lane -> 2x ds_load_b128 each
    frag16 b0, b1;
    const uint4* bp0 = reinterpret_cast<const uint4*>(&BshT[l15 * LDA + hsel * 16]);
    b0.q[0] = bp0[0]; b0.q[1] = bp0[1];
    const uint4* bp1 = reinterpret_cast<const uint4*>(&BshT[(16 + l15) * LDA + hsel * 16]);
    b1.q[0] = bp1[0]; b1.q[1] = bp1[1];

    acc0 = wmma_f16(a.v, b0.v, acc0);
    acc1 = wmma_f16(a.v, b1.v, acc1);
    __syncthreads();
  }

  int c0 = bn + l15, c1 = bn + 16 + l15;
#pragma unroll
  for (int r = 0; r < 8; ++r) {
    int row = bm + wid * 16 + r + hsel * 8;
    float v0 = acc0[r] + bias[c0];
    float v1 = acc1[r] + bias[c1];
    if (resid) {
      v0 += resid[(size_t)row * N + c0];
      v1 += resid[(size_t)row * N + c1];
    }
    if (relu) { v0 = fmaxf(v0, 0.f); v1 = fmaxf(v1, 0.f); }
    C[(size_t)row * N + c0] = v0;
    C[(size_t)row * N + c1] = v1;
  }
}

// ---------------------------------------------------------------------------
// Flash attention: one wave per (b, h, 16-query tile). hd=32 -> WMMA K=32.
// grid = (S_PAD/16, B*H), block = 32.
// ---------------------------------------------------------------------------
__global__ __launch_bounds__(32) void flash_attn(
    const float* __restrict__ Q, const float* __restrict__ Km,
    const float* __restrict__ V, float* __restrict__ O) {
  __shared__ alignas(16) _Float16 Psh[16 * 32];
  int lane = threadIdx.x;
  int hsel = lane >> 4, l15 = lane & 15;
  int qt = blockIdx.x;
  int b = blockIdx.y >> 3, h = blockIdx.y & 7;
  size_t base = ((size_t)b * S_PAD) * DM + h * 32;

  frag16 qa;  // Q tile as A fragment (16 queries x 32 hd)
  {
    const float* qr = Q + base + (size_t)(qt * 16 + l15) * DM;
    const float4* qa4 = reinterpret_cast<const float4*>(qr + hsel * 16);
    const float4* qb4 = reinterpret_cast<const float4*>(qr + hsel * 16 + 8);
    cvt8(qa, 0, qa4[0], qa4[1]);
    cvt8(qa, 8, qb4[0], qb4[1]);
  }
  float m[8], l[8];
  v8f o0 = {}, o1 = {};
#pragma unroll
  for (int r = 0; r < 8; ++r) { m[r] = -1e30f; l[r] = 0.f; }
  const float scale = 0.17677669529663687f;  // 1/sqrt(32)

  for (int kb = 0; kb < S_PAD; kb += 32) {
    // K^T fragments (B layout: 32 hd x 16 keys), two key tiles
    frag16 kf0, kf1;
    {
      const float4* k4a = reinterpret_cast<const float4*>(
          Km + base + (size_t)(kb + l15) * DM + hsel * 16);
      const float4* k4b = reinterpret_cast<const float4*>(
          Km + base + (size_t)(kb + 16 + l15) * DM + hsel * 16);
      cvt8(kf0, 0, k4a[0], k4a[1]);
      cvt8(kf0, 8, k4a[2], k4a[3]);
      cvt8(kf1, 0, k4b[0], k4b[1]);
      cvt8(kf1, 8, k4b[2], k4b[3]);
    }
    v8f z0 = {}, z1 = {};
    v8f s0 = wmma_f16(qa.v, kf0.v, z0);
    v8f s1 = wmma_f16(qa.v, kf1.v, z1);

    bool msk0 = (kb + l15) >= S_VALID;
    bool msk1 = (kb + 16 + l15) >= S_VALID;
    float p0[8], p1[8], rm[8];
#pragma unroll
    for (int r = 0; r < 8; ++r) {
      p0[r] = msk0 ? -1e30f : s0[r] * scale;
      p1[r] = msk1 ? -1e30f : s1[r] * scale;
      rm[r] = fmaxf(p0[r], p1[r]);
    }
    // row-max across the 16 lanes holding the row's columns
#pragma unroll
    for (int r = 0; r < 8; ++r)
#pragma unroll
      for (int mk = 1; mk < 16; mk <<= 1)
        rm[r] = fmaxf(rm[r], __shfl_xor(rm[r], mk, 32));

    float alpha[8], rs[8];
#pragma unroll
    for (int r = 0; r < 8; ++r) {
      float mn = fmaxf(m[r], rm[r]);
      alpha[r] = __expf(m[r] - mn);
      p0[r] = __expf(p0[r] - mn);
      p1[r] = __expf(p1[r] - mn);
      rs[r] = p0[r] + p1[r];
      m[r] = mn;
    }
#pragma unroll
    for (int r = 0; r < 8; ++r) {
#pragma unroll
      for (int mk = 1; mk < 16; mk <<= 1) rs[r] += __shfl_xor(rs[r], mk, 32);
      l[r] = l[r] * alpha[r] + rs[r];
      o0[r] *= alpha[r];
      o1[r] *= alpha[r];
    }
    // re-layout P (D layout -> A layout) through LDS
    __syncthreads();
#pragma unroll
    for (int r = 0; r < 8; ++r) {
      int prow = r + hsel * 8;
      Psh[prow * 32 + l15] = (_Float16)p0[r];
      Psh[prow * 32 + 16 + l15] = (_Float16)p1[r];
    }
    __syncthreads();
    frag16 pa;
    pa.q[0] = *reinterpret_cast<const uint4*>(&Psh[l15 * 32 + hsel * 16]);
    pa.q[1] = *reinterpret_cast<const uint4*>(&Psh[l15 * 32 + hsel * 16 + 8]);

    // V fragments (32 keys x 16 hd-cols), two N halves (strided gather)
    frag16 vf0, vf1;
#pragma unroll
    for (int j = 0; j < 16; ++j) {
      const float* vr = V + base + (size_t)(kb + hsel * 16 + j) * DM;
      vf0.h[j] = (_Float16)vr[l15];
      vf1.h[j] = (_Float16)vr[16 + l15];
    }
    o0 = wmma_f16(pa.v, vf0.v, o0);
    o1 = wmma_f16(pa.v, vf1.v, o1);
  }

#pragma unroll
  for (int r = 0; r < 8; ++r) {
    int orow = qt * 16 + r + hsel * 8;
    float inv = 1.0f / l[r];
    O[base + (size_t)orow * DM + l15] = o0[r] * inv;
    O[base + (size_t)orow * DM + 16 + l15] = o1[r] * inv;
  }
}

// ---------------------------------------------------------------------------
// LayerNorm over 256 cols, one wave per row; block 256 => 8 rows/block
// ---------------------------------------------------------------------------
__global__ __launch_bounds__(256) void ln_kernel(
    const float* __restrict__ in, const float* __restrict__ g,
    const float* __restrict__ bb, float* __restrict__ out) {
  int tid = threadIdx.x, lane = tid & 31, wid = tid >> 5;
  size_t row = (size_t)blockIdx.x * 8 + wid;
  const float* xr = in + row * 256;
  float v[8], s = 0.f;
#pragma unroll
  for (int i = 0; i < 8; ++i) { v[i] = xr[lane + 32 * i]; s += v[i]; }
#pragma unroll
  for (int mk = 1; mk < 32; mk <<= 1) s += __shfl_xor(s, mk, 32);
  float mu = s * (1.0f / 256.0f);
  float s2 = 0.f;
#pragma unroll
  for (int i = 0; i < 8; ++i) { float d = v[i] - mu; s2 += d * d; }
#pragma unroll
  for (int mk = 1; mk < 32; mk <<= 1) s2 += __shfl_xor(s2, mk, 32);
  float rinv = rsqrtf(s2 * (1.0f / 256.0f) + 1e-5f);
  float* orow = out + row * 256;
#pragma unroll
  for (int i = 0; i < 8; ++i) {
    int c = lane + 32 * i;
    orow[c] = (v[i] - mu) * rinv * g[c] + bb[c];
  }
}

// ---------------------------------------------------------------------------
// x = emb + pos(pe on the fly) + ct, with CLS at s=0 and zero padding rows
// grid = B*S_PAD rows, block 256 (one col/thread)
// ---------------------------------------------------------------------------
__global__ __launch_bounds__(256) void embed_combine(
    const float* __restrict__ emb, const float* __restrict__ ct,
    const int* __restrict__ xc, const int* __restrict__ yc,
    const float* __restrict__ cls, float* __restrict__ X) {
  int col = threadIdx.x;
  int row = blockIdx.x;
  int b = row / S_PAD, s = row % S_PAD;
  float val;
  if (s == 0) {
    val = cls[col];
  } else if (s <= 1024) {
    int tok = b * 1024 + (s - 1);
    int p = (col < 128) ? xc[tok] : yc[tok];
    p = p < 0 ? 0 : (p > 999 ? 999 : p);
    int j = col & 127;
    int t = j >> 1;
    // div[t] = 10000^(-t/64)
    float dv = __expf(-(float)t * (9.210340371976184f / 64.0f));
    float ang = (float)p * dv;
    float pos = (j & 1) ? cosf(ang) : sinf(ang);
    val = emb[(size_t)tok * 256 + col] + ct[(size_t)tok * 256 + col] + pos;
  } else {
    val = 0.f;  // padding rows kept zero/finite
  }
  X[(size_t)row * 256 + col] = val;
}

// ---------------------------------------------------------------------------
// head: cls_out copy + relu(cls@Wh1+bh1)@Wh2+bh2 ; grid=8, block=256
// ---------------------------------------------------------------------------
__global__ __launch_bounds__(256) void head_kernel(
    const float* __restrict__ Xf, const float* __restrict__ Wh1,
    const float* __restrict__ bh1, const float* __restrict__ Wh2,
    const float* __restrict__ bh2, float* __restrict__ out) {
  __shared__ float cls[256], hid[256];
  int b = blockIdx.x, tid = threadIdx.x;
  float c = Xf[((size_t)b * S_PAD) * 256 + tid];
  cls[tid] = c;
  out[b * 256 + tid] = c;  // cls_out
  __syncthreads();
  float acc = bh1[tid];
  for (int k = 0; k < 256; ++k) acc += cls[k] * Wh1[k * 256 + tid];
  hid[tid] = fmaxf(acc, 0.f);
  __syncthreads();
  if (tid < 2) {
    float a2 = bh2[tid];
    for (int i = 0; i < 256; ++i) a2 += hid[i] * Wh2[i * 2 + tid];
    out[2048 + b * 2 + tid] = a2;  // logits
  }
}

// ---------------------------------------------------------------------------
extern "C" void kernel_launch(void* const* d_in, const int* in_sizes, int n_in,
                              void* d_out, int out_size, void* d_ws, size_t ws_size,
                              hipStream_t stream) {
  (void)in_sizes; (void)n_in; (void)out_size; (void)ws_size;
  const float* cf    = (const float*)d_in[0];
  const int*   xc    = (const int*)d_in[1];
  const int*   yc    = (const int*)d_in[2];
  const float* W_emb = (const float*)d_in[3];
  const float* b_emb = (const float*)d_in[4];
  const float* W_ct1 = (const float*)d_in[5];
  const float* b_ct1 = (const float*)d_in[6];
  const float* W_ct2 = (const float*)d_in[7];
  const float* b_ct2 = (const float*)d_in[8];
  const float* clsT  = (const float*)d_in[9];
  const float* Wq    = (const float*)d_in[10];
  const float* bq    = (const float*)d_in[11];
  const float* Wk    = (const float*)d_in[12];
  const float* bk    = (const float*)d_in[13];
  const float* Wv    = (const float*)d_in[14];
  const float* bv    = (const float*)d_in[15];
  const float* Wo    = (const float*)d_in[16];
  const float* bo    = (const float*)d_in[17];
  const float* ln1g  = (const float*)d_in[18];
  const float* ln1b  = (const float*)d_in[19];
  const float* ln2g  = (const float*)d_in[20];
  const float* ln2b  = (const float*)d_in[21];
  const float* Wf1   = (const float*)d_in[22];
  const float* bf1   = (const float*)d_in[23];
  const float* Wf2   = (const float*)d_in[24];
  const float* bf2   = (const float*)d_in[25];
  const float* lnfg  = (const float*)d_in[26];
  const float* lnfb  = (const float*)d_in[27];
  const float* Wh1   = (const float*)d_in[28];
  const float* bh1   = (const float*)d_in[29];
  const float* Wh2   = (const float*)d_in[30];
  const float* bh2   = (const float*)d_in[31];
  float* out = (float*)d_out;

  // workspace layout (floats)
  float* ws = (float*)d_ws;
  const size_t AF = (size_t)8448 * 256;  // one activation buffer (B*S_PAD x 256)
  float* X   = ws;
  float* Qb  = ws + AF;
  float* Kb  = ws + 2 * AF;
  float* Vb  = ws + 3 * AF;
  float* ATT = ws + 4 * AF;
  float* TMP = ws + 5 * AF;
  float* H1  = ws + 6 * AF;  // B*S_PAD x 1024
  _Float16* hw = (_Float16*)(ws + 6 * AF + (size_t)8448 * 1024);
  _Float16* hW_emb = hw;
  _Float16* hW_ct1 = hW_emb + 16384;
  _Float16* hW_ct2 = hW_ct1 + 8192;
  _Float16* hWq    = hW_ct2 + 32768;
  _Float16* hWk    = hWq + 262144;
  _Float16* hWv    = hWk + 262144;
  _Float16* hWo    = hWv + 262144;
  _Float16* hWf1   = hWo + 262144;
  _Float16* hWf2   = hWf1 + 1048576;

  auto CVT = [&](const float* s, _Float16* d, int n) {
    cvt_f32_f16<<<(n + 255) / 256, 256, 0, stream>>>(s, d, n);
  };
  CVT(W_emb, hW_emb, 16384);
  CVT(W_ct1, hW_ct1, 8192);
  CVT(W_ct2, hW_ct2, 32768);
  CVT(Wq, hWq, 262144);
  CVT(Wk, hWk, 262144);
  CVT(Wv, hWv, 262144);
  CVT(Wo, hWo, 262144);
  CVT(Wf1, hWf1, 1048576);
  CVT(Wf2, hWf2, 1048576);

  auto GEMM = [&](const float* A, const _Float16* W, const float* bias,
                  const float* resid, float* C, int M, int N, int K, int relu) {
    dim3 g(M / 128, N / 32);
    gemm_rrr<<<g, 256, 0, stream>>>(A, W, bias, resid, C, M, N, K, relu);
  };

  // embedding path (M = 8*1024 tokens)
  GEMM(cf, hW_emb, b_emb, nullptr, TMP, 8192, 256, 64, 0);
  GEMM(cf, hW_ct1, b_ct1, nullptr, H1, 8192, 128, 64, 1);
  GEMM(H1, hW_ct2, b_ct2, nullptr, ATT, 8192, 256, 128, 0);
  embed_combine<<<8 * S_PAD, 256, 0, stream>>>(TMP, ATT, xc, yc, clsT, X);

  const int Mr = 8 * S_PAD;  // 8448
  for (int lyr = 0; lyr < 4; ++lyr) {
    GEMM(X, hWq + lyr * 65536, bq + lyr * 256, nullptr, Qb, Mr, 256, 256, 0);
    GEMM(X, hWk + lyr * 65536, bk + lyr * 256, nullptr, Kb, Mr, 256, 256, 0);
    GEMM(X, hWv + lyr * 65536, bv + lyr * 256, nullptr, Vb, Mr, 256, 256, 0);
    flash_attn<<<dim3(S_PAD / 16, 64), 32, 0, stream>>>(Qb, Kb, Vb, ATT);
    GEMM(ATT, hWo + lyr * 65536, bo + lyr * 256, X, TMP, Mr, 256, 256, 0);
    ln_kernel<<<Mr / 8, 256, 0, stream>>>(TMP, ln1g + lyr * 256, ln1b + lyr * 256, X);
    GEMM(X, hWf1 + lyr * 262144, bf1 + lyr * 1024, nullptr, H1, Mr, 1024, 256, 1);
    GEMM(H1, hWf2 + lyr * 262144, bf2 + lyr * 256, X, TMP, Mr, 256, 1024, 0);
    ln_kernel<<<Mr / 8, 256, 0, stream>>>(TMP, ln2g + lyr * 256, ln2b + lyr * 256, X);
  }
  ln_kernel<<<Mr / 8, 256, 0, stream>>>(X, lnfg, lnfb, TMP);
  head_kernel<<<8, 256, 0, stream>>>(TMP, Wh1, bh1, Wh2, bh2, out);
}
